// DualAttention_89936615178479
// MI455X (gfx1250) — compile-verified, matrix operand order checked by reading, not run on hardware
//
#include <hip/hip_runtime.h>
#include <hip/hip_bf16.h>
#include <math.h>

#define Bn 1024
#define Ln 196
#define Hn 1024
#define An 512
#define TSTRIDE 36      // LDS row stride (bf16 elems): 72B rows -> 8B-aligned frags
#define KPAD_CH 224     // L=196 zero-padded to 7 K-tiles of 32

typedef __attribute__((ext_vector_type(16))) __bf16 v16bf;
typedef __attribute__((ext_vector_type(8)))  float  v8f;

union FragLDS { uint2 q[4]; v16bf v; };   // A fragment gathered from LDS (2 halves)
union FragGbl { uint4 q[2]; v16bf v; };   // B fragment: contiguous 32B from packed global

__device__ __forceinline__ unsigned short f2bf(float f) {
  unsigned int u = __float_as_uint(f);
  return (unsigned short)((u + 0x7FFFu + ((u >> 16) & 1u)) >> 16);  // RNE
}

// branchless tanh: avoids ocml's exec-mask branches in the fused epilogue
__device__ __forceinline__ float fast_tanh(float x) {
  x = fminf(15.0f, fmaxf(-15.0f, x));
  float e = __expf(2.0f * x);
  return (e - 1.0f) / (e + 1.0f);
}

// A-operand per ISA 7.12.2: elems 0..7 = K[kb..kb+7], 8..15 = K[kb+16..kb+23], kb=8*(lane>=16)
__device__ __forceinline__ v16bf load_fragA(const unsigned short* base) {
  FragLDS f;
  f.q[0] = *(const uint2*)(base + 0);
  f.q[1] = *(const uint2*)(base + 4);
  f.q[2] = *(const uint2*)(base + 16);
  f.q[3] = *(const uint2*)(base + 20);
  return f.v;
}

// ---- one-time: pack fp32 weight [K,A] into WMMA-fragment-ordered bf16 -------
// packed[((kt*32+nt)*32 + lane)*16 + e] = bf16(W[k,n]),
//   n = nt*16 + (lane&15), k = kt*32 + 8*(lane>=16) + e + (e>=8 ? 8 : 0)
__global__ void __launch_bounds__(256)
pack_w_kernel(const float* __restrict__ W, int K, unsigned short* __restrict__ out) {
  int idx  = blockIdx.x * 256 + threadIdx.x;
  int e    = idx & 15;
  int lane = (idx >> 4) & 31;
  int nt   = (idx >> 9) & 31;
  int kt   = idx >> 14;
  int k = kt * 32 + ((lane >> 4) << 3) + e + ((e >= 8) ? 8 : 0);
  int n = nt * 16 + (lane & 15);
  float v = (k < K) ? W[(size_t)k * An + n] : 0.0f;
  out[idx] = f2bf(v);
}

// ---------------- p_h = h @ W_h + b_h   [B,A] ----------------
__global__ void __launch_bounds__(256)
ph_kernel(const float* __restrict__ h, const float* __restrict__ W_h,
          const float* __restrict__ b_h, float* __restrict__ ph) {
  int a = blockIdx.x * 256 + threadIdx.x;
  int b = blockIdx.y;
  const float* hrow = h + (size_t)b * Hn;
  float acc = b_h[a];
  #pragma unroll 8
  for (int k = 0; k < Hn; ++k) acc = fmaf(hrow[k], W_h[(size_t)k * An + a], acc);
  ph[(size_t)b * An + a] = acc;
}

// ---------------- spatial: fused GEMM + tanh·w_alpha -> logits_s [B*L] -------
// M-block = 32 rows, N = 512 full; 8 waves = 2 m-tiles x 4 n-groups(128 cols)
__global__ void __launch_bounds__(256)
spatial_logits_kernel(const float* __restrict__ X, const unsigned short* __restrict__ Wp,
                      const float* __restrict__ b_att, const float* __restrict__ ph,
                      const float* __restrict__ w_alpha, const float* __restrict__ b_alpha,
                      float* __restrict__ logits) {
  __shared__ unsigned short lds_a[32 * TSTRIDE];
  __shared__ float lds_red[32];

  const int tid  = threadIdx.x;
  const int lane = tid & 31;
  const int wv   = tid >> 5;
  const int mt   = wv & 1;               // m-tile within block
  const int ng   = wv >> 1;              // n-group: 128 cols
  const int m0   = blockIdx.x * 32;
  const int kb   = (lane >> 4) << 3;

  if (tid < 32) lds_red[tid] = 0.0f;

  v8f acc[8] = {};

  for (int k0 = 0; k0 < Hn; k0 += 32) {
    __syncthreads();
    {   // stage A tile 32x32 (fp32 -> bf16), streaming NT loads
      int k  = tid & 31;
      int mi = tid >> 5;                 // 0..7
      #pragma unroll
      for (int r = 0; r < 4; ++r) {
        int mm = mi + r * 8;
        size_t gi = (size_t)(m0 + mm) * Hn + (k0 + k);
        if (k0 + 32 < Hn) __builtin_prefetch(&X[gi + 32], 0, 1);
        lds_a[mm * TSTRIDE + k] = f2bf(__builtin_nontemporal_load(&X[gi]));
      }
    }
    __syncthreads();
    v16bf af = load_fragA(lds_a + (mt * 16 + (lane & 15)) * TSTRIDE + kb);
    const int kt = k0 >> 5;
    #pragma unroll
    for (int tt = 0; tt < 8; ++tt) {
      int nt = ng * 8 + tt;
      const uint4* pb = (const uint4*)(Wp + (((size_t)kt * 32 + nt) * 32 + lane) * 16);
      FragGbl bf; bf.q[0] = pb[0]; bf.q[1] = pb[1];   // 2x global_load_b128, coalesced
      acc[tt] = __builtin_amdgcn_wmma_f32_16x16x32_bf16(false, af, false, bf.v,
                                                        (short)0, acc[tt], false, false);
    }
  }

  // fused epilogue: rsum[i] = sum_n tanh(C + p_h[b,n] + b_att[n]) * w_alpha[n]
  float rsum[8];
  #pragma unroll
  for (int i = 0; i < 8; ++i) rsum[i] = 0.0f;
  #pragma unroll
  for (int tt = 0; tt < 8; ++tt) {
    int n = ng * 128 + tt * 16 + (lane & 15);
    float ba = b_att[n];
    float wa = w_alpha[n];
    #pragma unroll
    for (int i = 0; i < 8; ++i) {
      int m = m0 + mt * 16 + i + kb;          // C/D layout: M = i + 8*(lane>=16)
      unsigned bb = (unsigned)m / 196u;       // tile may span batch boundary
      float v = acc[tt][i] + ph[(size_t)bb * An + n] + ba;
      rsum[i] += fast_tanh(v) * wa;
    }
  }
  __syncthreads();
  #pragma unroll
  for (int i = 0; i < 8; ++i)
    atomicAdd(&lds_red[mt * 16 + i + kb], rsum[i]);   // ds_add_f32
  __syncthreads();
  if (tid < 32) logits[m0 + tid] = lds_red[tid] + b_alpha[0];
}

// ---------------- channel: fused GEMM + tanh·w_beta -> logits_c [B,H] --------
// per-batch: C[hh,n] = sum_l X[b,l,hh] * Wch[l,n];  M=1024 (32/block), K=196 pad 224
__global__ void __launch_bounds__(256)
channel_logits_kernel(const float* __restrict__ X, const unsigned short* __restrict__ Wp,
                      const float* __restrict__ b_ch, const float* __restrict__ ph,
                      const float* __restrict__ w_beta, const float* __restrict__ b_beta,
                      float* __restrict__ logits) {
  __shared__ unsigned short lds_a[32 * TSTRIDE];
  __shared__ float lds_red[32];

  const int tid  = threadIdx.x;
  const int lane = tid & 31;
  const int wv   = tid >> 5;
  const int mt   = wv & 1;
  const int ng   = wv >> 1;
  const int b    = blockIdx.y;
  const int m0   = blockIdx.x * 32;              // hh base
  const int kb   = (lane >> 4) << 3;
  const float* Xb  = X + (size_t)b * Ln * Hn;
  const float* phb = ph + (size_t)b * An;

  if (tid < 32) lds_red[tid] = 0.0f;

  v8f acc[8] = {};

  for (int k0 = 0; k0 < KPAD_CH; k0 += 32) {
    __syncthreads();
    {   // stage A = X^T tile: lds_a[hh][l], transpose on LDS write, zero-pad l>=196
      int hh = tid & 31;
      int lb = tid >> 5;                 // 0..7
      #pragma unroll
      for (int r = 0; r < 4; ++r) {
        int ll = lb + r * 8;
        int gl = k0 + ll;
        float v = (gl < Ln) ? __builtin_nontemporal_load(&Xb[(size_t)gl * Hn + m0 + hh]) : 0.0f;
        lds_a[hh * TSTRIDE + ll] = f2bf(v);
      }
    }
    __syncthreads();
    v16bf af = load_fragA(lds_a + (mt * 16 + (lane & 15)) * TSTRIDE + kb);
    const int kt = k0 >> 5;
    #pragma unroll
    for (int tt = 0; tt < 8; ++tt) {
      int nt = ng * 8 + tt;
      const uint4* pb = (const uint4*)(Wp + (((size_t)kt * 32 + nt) * 32 + lane) * 16);
      FragGbl bf; bf.q[0] = pb[0]; bf.q[1] = pb[1];
      acc[tt] = __builtin_amdgcn_wmma_f32_16x16x32_bf16(false, af, false, bf.v,
                                                        (short)0, acc[tt], false, false);
    }
  }

  float rsum[8];
  #pragma unroll
  for (int i = 0; i < 8; ++i) rsum[i] = 0.0f;
  #pragma unroll
  for (int tt = 0; tt < 8; ++tt) {
    int n = ng * 128 + tt * 16 + (lane & 15);
    float pb2 = phb[n] + b_ch[n];
    float wb  = w_beta[n];
    #pragma unroll
    for (int i = 0; i < 8; ++i)
      rsum[i] += fast_tanh(acc[tt][i] + pb2) * wb;
  }
  __syncthreads();
  #pragma unroll
  for (int i = 0; i < 8; ++i)
    atomicAdd(&lds_red[mt * 16 + i + kb], rsum[i]);
  __syncthreads();
  if (tid < 32) logits[(size_t)b * Hn + m0 + tid] = lds_red[tid] + b_beta[0];
}

// ---- fused per-batch: both softmaxes + single pass over X[b] for both -------
// weighted reductions.  1024 threads = 32 waves; thread t owns column hh = t.
// weighted_channel uses per-row wave shfl-reduction -> 32 partials in LDS,
// reduced once after the streaming loop (no per-row __syncthreads).
__global__ void __launch_bounds__(1024)
fused_attend_kernel(const float* __restrict__ X,
                    const float* __restrict__ logits_s,
                    const float* __restrict__ logits_c,
                    float* __restrict__ out_ws,     // [B,H] weighted_spatial
                    float* __restrict__ out_wc,     // [B,L] weighted_channel
                    float* __restrict__ out_wspa) { // [B,L] w_spatial
  __shared__ float red[1024];
  __shared__ float sw[Ln];
  __shared__ float wpart[Ln * 32];

  const int b = blockIdx.x;
  const int t = threadIdx.x;
  const int lane = t & 31;
  const int wv   = t >> 5;

  // softmax over L=196 -> sw[], also emit w_spatial output
  float vs = (t < Ln) ? logits_s[(size_t)b * Ln + t] : -INFINITY;
  red[t] = vs; __syncthreads();
  for (int s = 512; s > 0; s >>= 1) { if (t < s) red[t] = fmaxf(red[t], red[t + s]); __syncthreads(); }
  float mx = red[0]; __syncthreads();
  float es = (t < Ln) ? __expf(vs - mx) : 0.0f;
  red[t] = es; __syncthreads();
  for (int s = 512; s > 0; s >>= 1) { if (t < s) red[t] += red[t + s]; __syncthreads(); }
  float inv = 1.0f / red[0];
  if (t < Ln) { float w = es * inv; sw[t] = w; out_wspa[(size_t)b * Ln + t] = w; }
  __syncthreads();

  // softmax over H=1024 -> per-thread channel weight wch
  float vc = logits_c[(size_t)b * Hn + t];
  red[t] = vc; __syncthreads();
  for (int s = 512; s > 0; s >>= 1) { if (t < s) red[t] = fmaxf(red[t], red[t + s]); __syncthreads(); }
  mx = red[0]; __syncthreads();
  float ec = __expf(vc - mx);
  red[t] = ec; __syncthreads();
  for (int s = 512; s > 0; s >>= 1) { if (t < s) red[t] += red[t + s]; __syncthreads(); }
  float wch = ec / red[0];
  __syncthreads();

  // single streaming pass over X[b] (NT: no reuse, keep L2 clean)
  const float* Xb = X + (size_t)b * Ln * Hn;
  float acc = 0.0f;
  for (int l = 0; l < Ln; ++l) {
    const float* px = &Xb[(size_t)l * Hn + t];
    if (l + 4 < Ln) __builtin_prefetch(px + 4 * Hn, 0, 1);
    float x = __builtin_nontemporal_load(px);
    acc = fmaf(sw[l], x, acc);                 // weighted_spatial accum (register)
    float p = wch * x;                         // weighted_channel row partial
    #pragma unroll
    for (int s2 = 16; s2 > 0; s2 >>= 1) p += __shfl_xor(p, s2, 32);
    if (lane == 0) wpart[l * 32 + wv] = p;
  }
  out_ws[(size_t)b * Hn + t] = acc;
  __syncthreads();
  if (t < Ln) {
    float s = 0.0f;
    #pragma unroll
    for (int i = 0; i < 32; ++i) s += wpart[t * 32 + i];
    out_wc[(size_t)b * Ln + t] = s;
  }
}

extern "C" void kernel_launch(void* const* d_in, const int* in_sizes, int n_in,
                              void* d_out, int out_size, void* d_ws, size_t ws_size,
                              hipStream_t stream) {
  const float* X       = (const float*)d_in[0];   // [B,L,H]
  const float* h       = (const float*)d_in[1];   // [B,H]
  const float* W_att   = (const float*)d_in[2];   // [H,A]
  const float* b_att   = (const float*)d_in[3];   // [A]
  const float* W_h     = (const float*)d_in[4];   // [H,A]
  const float* b_h     = (const float*)d_in[5];   // [A]
  const float* w_alpha = (const float*)d_in[6];   // [A]
  const float* b_alpha = (const float*)d_in[7];   // scalar
  const float* W_ch    = (const float*)d_in[8];   // [L,A]
  const float* b_ch    = (const float*)d_in[9];   // [A]
  const float* w_beta  = (const float*)d_in[10];  // [A]
  const float* b_beta  = (const float*)d_in[11];  // scalar

  // workspace: p_h [B*A] | logits_s [B*L] | logits_c [B*H]
  //            | packed W_att bf16 [1024*512] | packed W_ch bf16 [224*512]
  float* ph       = (float*)d_ws;
  float* logits_s = ph + (size_t)Bn * An;
  float* logits_c = logits_s + (size_t)Bn * Ln;
  unsigned short* Wp_att = (unsigned short*)(logits_c + (size_t)Bn * Hn);
  unsigned short* Wp_ch  = Wp_att + (size_t)Hn * An;

  // outputs: weighted_spatial [B*H] | weighted_channel [B*L] | w_spatial [B*L]
  float* out_wsum = (float*)d_out;
  float* out_wcha = out_wsum + (size_t)Bn * Hn;
  float* out_wspa = out_wcha + (size_t)Bn * Ln;

  pack_w_kernel<<<(Hn * An) / 256, 256, 0, stream>>>(W_att, Hn, Wp_att);
  pack_w_kernel<<<(KPAD_CH * An) / 256, 256, 0, stream>>>(W_ch, Ln, Wp_ch);

  ph_kernel<<<dim3(An / 256, Bn), 256, 0, stream>>>(h, W_h, b_h, ph);

  spatial_logits_kernel<<<(Bn * Ln) / 32, 256, 0, stream>>>(
      X, Wp_att, b_att, ph, w_alpha, b_alpha, logits_s);

  channel_logits_kernel<<<dim3(Hn / 32, Bn), 256, 0, stream>>>(
      X, Wp_ch, b_ch, ph, w_beta, b_beta, logits_c);

  fused_attend_kernel<<<Bn, 1024, 0, stream>>>(
      X, logits_s, logits_c, out_wsum, out_wcha, out_wspa);
}
